// SparseGaussianProcess_8100308321081
// MI455X (gfx1250) — matrix operand order repaired and checked by reading, not compiled, not run on hardware
//
#include <hip/hip_runtime.h>

typedef __attribute__((ext_vector_type(16))) _Float16 v16h;
typedef __attribute__((ext_vector_type(8)))  _Float16 v8h;
typedef __attribute__((ext_vector_type(8)))  float    v8f;
typedef __attribute__((ext_vector_type(4)))  float    v4f;
typedef __attribute__((ext_vector_type(2)))  float    v2f;

#if __has_builtin(__builtin_amdgcn_wmma_f32_16x16x4_f32)
#define HAVE_WMMA_F32X4 1
#else
#define HAVE_WMMA_F32X4 0
#endif

namespace {
constexpr int kN      = 65536;
constexpr int kID     = 16;
constexpr int kOD     = 8;
constexpr int kM      = 1024;
constexpr int kS      = 8;
constexpr int kL      = 64;
constexpr int kC      = kS * kOD;       // 64 fused output columns (s,o)
constexpr int kKA     = kM + kOD * kL;  // 1536 augmented reduction dim
constexpr int kWP     = kKA + 8;        // W_aug LDS row pitch (halves): 3088 B
                                        // -> lane stride 4 banks, conflict-free
constexpr int kSP     = 40;             // staging row pitch (halves): 80 B
                                        // -> lane stride 20 banks, conflict-free
constexpr int kWaves  = 16;             // waves per workgroup (wave32)
constexpr int kRowsWG = kWaves * 16;    // 256 x-rows per workgroup
constexpr float kBW   = 0.17677669529663689f; // sqrt(2/64)
} // namespace

__device__ __forceinline__ v8f vzero8() {
  v8f z;
  #pragma unroll
  for (int i = 0; i < 8; ++i) z[i] = 0.0f;
  return z;
}

__global__ __launch_bounds__(kWaves * 32, 1)
void sgp_fused_kernel(const float* __restrict__ x,    // [N,16]
                      const float* __restrict__ z,    // [1024,16]
                      const float* __restrict__ iw,   // [8,8,1024]
                      const float* __restrict__ pf,   // [8,16,64]
                      const float* __restrict__ pph,  // [8,64]
                      const float* __restrict__ pw,   // [8,8,64]
                      float* __restrict__ out)        // [64,N]
{
  extern __shared__ float smem[];
  float*    z2    = smem;                              // 1024 f32
  float*    xsq   = smem + kM;                         // 16 waves * 16 f32
  _Float16* waug  = (_Float16*)(xsq + kWaves * 16);    // 64 * kWP f16
  _Float16* stage = waug + kC * kWP;                   // 16 waves * 16*kSP f16

  const int tid  = threadIdx.x;
  const int lane = tid & 31;
  const int wv   = tid >> 5;
  const int n15  = lane & 15;
  const int hl   = lane >> 4;      // half-wave select
  const int offA = hl * 2;         // K sub-offset for f32 WMMA frags

  // ---- cooperative: z row norms ----
  for (int m = tid; m < kM; m += kWaves * 32) {
    const float* zr = z + m * kID;
    float s = 0.f;
    #pragma unroll
    for (int k = 0; k < kID; ++k) s += zr[k] * zr[k];
    z2[m] = s;
  }
  // ---- cooperative: augmented weight matrix, f16, [c][k] with padded pitch ----
  for (int c = 0; c < kC; ++c) {
    _Float16* wrow = waug + c * kWP;
    const float* iwr = iw + c * kM;
    const float* pwr = pw + c * kL;
    const int oc = c & 7;
    for (int k = tid; k < kKA; k += kWaves * 32) {
      float v;
      if (k < kM) {
        v = iwr[k];                                  // data weights
      } else {
        const int j = k - kM;                        // prior feature (o,l)
        v = ((j >> 6) == oc) ? kBW * pwr[j & 63] : 0.0f;
      }
      wrow[k] = (_Float16)v;
    }
  }
  __syncthreads();

  const int n0 = blockIdx.x * kRowsWG + wv * 16;  // this wave's 16 x-rows

  // ---- per-wave: x row norms via LDS (DS ops are in-order per wave) ----
  if (lane < 16) {
    const float* xr = x + (size_t)(n0 + lane) * kID;
    float s = 0.f;
    #pragma unroll
    for (int k = 0; k < kID; ++k) s += xr[k] * xr[k];
    xsq[wv * 16 + lane] = s;
  }
  __builtin_amdgcn_wave_barrier();
  float x2v[8];
  #pragma unroll
  for (int v = 0; v < 8; ++v) x2v[v] = xsq[wv * 16 + v + 8 * hl];

  // ---- x-tile A fragments for v_wmma_f32_16x16x4_f32 (4 chunks of K=4) ----
  v2f ax[4];
  {
    const float* xr = x + (size_t)(n0 + n15) * kID;
    #pragma unroll
    for (int t = 0; t < 4; ++t) ax[t] = *(const v2f*)(xr + t * 4 + offA);
  }

  _Float16* stw = stage + wv * (16 * kSP);

  v8f accO[4];
  #pragma unroll
  for (int ct = 0; ct < 4; ++ct) accO[ct] = vzero8();

  // ================= main K loop: 48 chunks of 32 features =================
  for (int f0 = 0; f0 < kKA; f0 += 32) {
    #pragma unroll
    for (int sub = 0; sub < 2; ++sub) {
      const int fb = f0 + sub * 16;
      v8f acc = vzero8();
      if (fb < kM) {
        // -------- RBF kernel features: exp(-0.5*||x-z||^2) --------
        const int m = fb + n15;
        const float* zr = z + m * kID;
#if HAVE_WMMA_F32X4
        #pragma unroll
        for (int t = 0; t < 4; ++t) {
          const v2f b = *(const v2f*)(zr + t * 4 + offA);
          acc = __builtin_amdgcn_wmma_f32_16x16x4_f32(false, ax[t], false, b,
                                                      (short)0, acc, false, false);
        }
#else
        #pragma unroll
        for (int v = 0; v < 8; ++v) {
          const float* xr = x + (size_t)(n0 + v + 8 * hl) * kID;
          float s = 0.f;
          for (int k = 0; k < kID; ++k) s += xr[k] * zr[k];
          acc[v] = s;
        }
#endif
        const float zz = z2[m];
        #pragma unroll
        for (int v = 0; v < 8; ++v) {
          float sq = x2v[v] + zz - 2.0f * acc[v];
          sq = fmaxf(sq, 0.0f);
          stw[(v + 8 * hl) * kSP + sub * 16 + n15] = (_Float16)__expf(-0.5f * sq);
        }
      } else {
        // -------- random Fourier features: cos(x.F + phase) --------
        const int j = (fb - kM) + n15;
        const int o = j >> 6;
        const int l = j & 63;
        const float* fc = pf + (size_t)(o * kID) * kL + l;
#if HAVE_WMMA_F32X4
        #pragma unroll
        for (int t = 0; t < 4; ++t) {
          v2f b;
          b.x = fc[(t * 4 + offA) * kL];
          b.y = fc[(t * 4 + offA + 1) * kL];
          acc = __builtin_amdgcn_wmma_f32_16x16x4_f32(false, ax[t], false, b,
                                                      (short)0, acc, false, false);
        }
#else
        #pragma unroll
        for (int v = 0; v < 8; ++v) {
          const float* xr = x + (size_t)(n0 + v + 8 * hl) * kID;
          float s = 0.f;
          for (int k = 0; k < kID; ++k) s += xr[k] * fc[k * kL];
          acc[v] = s;
        }
#endif
        const float ph = pph[o * kL + l];
        #pragma unroll
        for (int v = 0; v < 8; ++v) {
          stw[(v + 8 * hl) * kSP + sub * 16 + n15] = (_Float16)__cosf(acc[v] + ph);
        }
      }
    }
    __builtin_amdgcn_wave_barrier();

    // A fragment (16x32 f16): lane<16 -> K 0..7 & 16..23 ; lane>=16 -> +8
    const int K0 = hl * 8;
    const v8h alo = *(const v8h*)(stw + n15 * kSP + K0);
    const v8h ahi = *(const v8h*)(stw + n15 * kSP + K0 + 16);
    const v16h a = __builtin_shufflevector(alo, ahi,
        0, 1, 2, 3, 4, 5, 6, 7, 8, 9, 10, 11, 12, 13, 14, 15);

    // preload all 4 B fragments (separate registers), then back-to-back WMMAs
    v16h bfr[4];
    #pragma unroll
    for (int ct = 0; ct < 4; ++ct) {
      const _Float16* wp = waug + (size_t)(ct * 16 + n15) * kWP + f0 + hl * 8;
      const v8h blo = *(const v8h*)(wp);
      const v8h bhi = *(const v8h*)(wp + 16);
      bfr[ct] = __builtin_shufflevector(blo, bhi,
          0, 1, 2, 3, 4, 5, 6, 7, 8, 9, 10, 11, 12, 13, 14, 15);
    }
    #pragma unroll
    for (int ct = 0; ct < 4; ++ct) {
      accO[ct] = __builtin_amdgcn_wmma_f32_16x16x32_f16(false, a, false, bfr[ct],
                                                        (short)0, accO[ct],
                                                        false, false);
    }
    __builtin_amdgcn_wave_barrier();
  }

  // ---- store: lane owns column c, 8 consecutive n values ----
  #pragma unroll
  for (int ct = 0; ct < 4; ++ct) {
    const int c = ct * 16 + n15;
    float* op = out + (size_t)c * kN + n0 + 8 * hl;
    v4f lo = {accO[ct][0], accO[ct][1], accO[ct][2], accO[ct][3]};
    v4f hv = {accO[ct][4], accO[ct][5], accO[ct][6], accO[ct][7]};
    *(v4f*)(op)     = lo;
    *(v4f*)(op + 4) = hv;
  }
}

extern "C" void kernel_launch(void* const* d_in, const int* in_sizes, int n_in,
                              void* d_out, int out_size, void* d_ws, size_t ws_size,
                              hipStream_t stream) {
  (void)in_sizes; (void)n_in; (void)out_size; (void)d_ws; (void)ws_size;
  const float* x   = (const float*)d_in[0];
  const float* z   = (const float*)d_in[1];
  const float* iw  = (const float*)d_in[2];
  const float* pf  = (const float*)d_in[3];
  const float* pph = (const float*)d_in[4];
  const float* pw  = (const float*)d_in[5];
  float* out = (float*)d_out;

  const size_t lds = (size_t)(kM + kWaves * 16) * sizeof(float)
                   + (size_t)(kC * kWP + kWaves * 16 * kSP) * sizeof(_Float16);
  // ~218 KB dynamic LDS (<= 320 KB per WGP on CDNA5)
  (void)hipFuncSetAttribute((const void*)sgp_fused_kernel,
                            hipFuncAttributeMaxDynamicSharedMemorySize,
                            (int)lds);

  dim3 grid(kN / kRowsWG);   // 256 workgroups
  dim3 block(kWaves * 32);   // 512 threads = 16 wave32
  sgp_fused_kernel<<<grid, block, lds, stream>>>(x, z, iw, pf, pph, pw, out);
}
